// Tacotron2_29626684408462
// MI455X (gfx1250) — matrix operand order repaired and checked by reading, not compiled, not run on hardware
//
#include <hip/hip_runtime.h>
#include <math.h>
#include <stdint.h>

// ---------------- CDNA5 WMMA types (wave32) ----------------
typedef __attribute__((ext_vector_type(16))) __bf16 v16bf;
typedef __attribute__((ext_vector_type(8)))  float  v8f;

struct __attribute__((aligned(16))) BF8   { __bf16 v[8];  };
struct __attribute__((aligned(16))) BF16x { __bf16 v[16]; };

static constexpr int NB = 64, NS = 512, NT = 1000, NMEL = 80;
static constexpr int DH = 1024;       // decoder hidden
static constexpr int AKP = 608;       // 592 (80 mel + 512 enc) padded to 19*32

// ---------------- helpers ----------------
__device__ __forceinline__ float sigf(float x) { return 1.f / (1.f + __expf(-x)); }
__device__ __forceinline__ float tanh_f(float x) {
  x = fminf(15.f, fmaxf(-15.f, x));
  float e = __expf(2.f * x);
  return (e - 1.f) / (e + 1.f);
}

// A fragment (16x32 bf16): lane L<16 holds row M=L, K = k0+0..15;
// lane L>=16 holds row M=L-16, K = k0+16..31.
__device__ __forceinline__ v16bf load_frag_a(const __bf16* base, int ld, int row, int k0, int lane) {
  const __bf16* p = base + (size_t)row * ld + k0 + ((lane >> 4) << 4);
  BF8 lo = *reinterpret_cast<const BF8*>(p);
  BF8 hi = *reinterpret_cast<const BF8*>(p + 8);
  v16bf r;
#pragma unroll
  for (int i = 0; i < 8; ++i) { r[i] = lo.v[i]; r[i + 8] = hi.v[i]; }
  return r;
}

// B fragment (32x16 bf16), B[k][n] = W[n][k]: lane L<16 holds col n=L, K=k0+0..15;
// lane L>=16 holds col n=L-16, K=k0+16..31.  W rows are contiguous in K.
__device__ __forceinline__ v16bf load_frag_b(const __bf16* W, int ld, int ncol, int k0, int lane) {
  const __bf16* p = W + (size_t)ncol * ld + k0 + ((lane >> 4) << 4);
  BF16x t = *reinterpret_cast<const BF16x*>(p);
  v16bf r;
#pragma unroll
  for (int i = 0; i < 16; ++i) r[i] = t.v[i];
  return r;
}

__device__ __forceinline__ v8f wmma_bf16(v16bf a, v16bf b, v8f c) {
  return __builtin_amdgcn_wmma_f32_16x16x32_bf16(false, a, false, b, (short)0, c, false, false);
}

// ---------------- fp32 -> bf16 weight conversion ----------------
__global__ void f2bf(const float* __restrict__ s, __bf16* __restrict__ d, size_t n) {
  size_t i = (size_t)blockIdx.x * blockDim.x + threadIdx.x;
  size_t st = (size_t)gridDim.x * blockDim.x;
  for (; i < n; i += st) d[i] = (__bf16)s[i];
}
__global__ void f2bf_pad(const float* __restrict__ s, __bf16* __restrict__ d, int rows, int K, int Kp) {
  size_t n = (size_t)rows * Kp;
  size_t i = (size_t)blockIdx.x * blockDim.x + threadIdx.x;
  size_t st = (size_t)gridDim.x * blockDim.x;
  for (; i < n; i += st) {
    int k = (int)(i % Kp), r = (int)(i / Kp);
    d[i] = (k < K) ? (__bf16)s[(size_t)r * K + k] : (__bf16)0.f;
  }
}

// ---------------- generic WMMA GEMM: C[M,N] = A[M,K] @ W[N,K]^T (+bias) ----------------
// 1M x 4N register blocking: one A fragment reused across 4 B tiles (A is the
// dominant stream at M=32768 -> 4x less activation L2 traffic). N must be %64.
__global__ void __launch_bounds__(256) gemm_bf16(
    const __bf16* __restrict__ A, int lda, const __bf16* __restrict__ W, int ldw,
    float* __restrict__ Cf, __bf16* __restrict__ Cbf, int ldc,
    const float* __restrict__ bias, int M, int N, int K) {
  const int lane = threadIdx.x & 31;
  const int wave = (blockIdx.x << 3) + (threadIdx.x >> 5);
  const int nwaves = gridDim.x << 3;
  const int tm = M >> 4, tnb = N >> 6;
  const int acol = lane & 15;
  for (int unit = wave; unit < tm * tnb; unit += nwaves) {
    const int mt = unit % tm, nb = unit / tm;
    const int m0 = mt << 4, n0 = nb << 6;
    v8f acc0, acc1, acc2, acc3;
#pragma unroll
    for (int r = 0; r < 8; ++r) {
      acc0[r] = bias ? bias[n0 + acol] : 0.f;
      acc1[r] = bias ? bias[n0 + 16 + acol] : 0.f;
      acc2[r] = bias ? bias[n0 + 32 + acol] : 0.f;
      acc3[r] = bias ? bias[n0 + 48 + acol] : 0.f;
    }
    for (int k0 = 0; k0 < K; k0 += 32) {
      v16bf a = load_frag_a(A, lda, m0 + acol, k0, lane);
      acc0 = wmma_bf16(a, load_frag_b(W, ldw, n0 +      acol, k0, lane), acc0);
      acc1 = wmma_bf16(a, load_frag_b(W, ldw, n0 + 16 + acol, k0, lane), acc1);
      acc2 = wmma_bf16(a, load_frag_b(W, ldw, n0 + 32 + acol, k0, lane), acc2);
      acc3 = wmma_bf16(a, load_frag_b(W, ldw, n0 + 48 + acol, k0, lane), acc3);
    }
    const int r0 = m0 + ((lane >> 4) << 3);
    if (Cf) {
#pragma unroll
      for (int r = 0; r < 8; ++r) {
        float* row = Cf + (size_t)(r0 + r) * ldc + n0 + acol;
        row[0] = acc0[r]; row[16] = acc1[r]; row[32] = acc2[r]; row[48] = acc3[r];
      }
    } else {
#pragma unroll
      for (int r = 0; r < 8; ++r) {
        __bf16* row = Cbf + (size_t)(r0 + r) * ldc + n0 + acol;
        row[0] = (__bf16)acc0[r]; row[16] = (__bf16)acc1[r];
        row[32] = (__bf16)acc2[r]; row[48] = (__bf16)acc3[r];
      }
    }
  }
}

// M=64 fused dual-GEMM for the decoder: gates = A1@W1^T + A2@W2^T + b.
// 4M x 1N blocking: one weight fragment feeds all four M-tiles -> each weight
// byte crosses L2 exactly once per step (weights are the per-step roofline term).
// global_prefetch pulls the weight row 128 elements (4 K blocks) ahead while WMMAs drain.
__device__ __forceinline__ void mm64_tiles(
    const __bf16* A1, int lda1, const __bf16* W1, int ldw1, int K1,
    const __bf16* A2, int lda2, const __bf16* W2, int ldw2, int K2,
    const float* bias, float* C, int N, int wave, int nwaves, int lane) {
  const int tn = N >> 4;
  const int acol = lane & 15;
  for (int nt = wave; nt < tn; nt += nwaves) {
    const int n0 = nt << 4;
    const float b0 = bias ? bias[n0 + acol] : 0.f;
    v8f acc0, acc1, acc2, acc3;
#pragma unroll
    for (int r = 0; r < 8; ++r) { acc0[r] = b0; acc1[r] = b0; acc2[r] = b0; acc3[r] = b0; }
    for (int k0 = 0; k0 < K1; k0 += 32) {
      v16bf bf = load_frag_b(W1, ldw1, n0 + acol, k0, lane);
      __builtin_prefetch((const void*)(W1 + (size_t)(n0 + acol) * ldw1 + k0 + 128), 0, 1);
      acc0 = wmma_bf16(load_frag_a(A1, lda1,      acol, k0, lane), bf, acc0);
      acc1 = wmma_bf16(load_frag_a(A1, lda1, 16 + acol, k0, lane), bf, acc1);
      acc2 = wmma_bf16(load_frag_a(A1, lda1, 32 + acol, k0, lane), bf, acc2);
      acc3 = wmma_bf16(load_frag_a(A1, lda1, 48 + acol, k0, lane), bf, acc3);
    }
    for (int k0 = 0; k0 < K2; k0 += 32) {
      v16bf bf = load_frag_b(W2, ldw2, n0 + acol, k0, lane);
      __builtin_prefetch((const void*)(W2 + (size_t)(n0 + acol) * ldw2 + k0 + 128), 0, 1);
      acc0 = wmma_bf16(load_frag_a(A2, lda2,      acol, k0, lane), bf, acc0);
      acc1 = wmma_bf16(load_frag_a(A2, lda2, 16 + acol, k0, lane), bf, acc1);
      acc2 = wmma_bf16(load_frag_a(A2, lda2, 32 + acol, k0, lane), bf, acc2);
      acc3 = wmma_bf16(load_frag_a(A2, lda2, 48 + acol, k0, lane), bf, acc3);
    }
    const int col = n0 + acol;
    const int rr = (lane >> 4) << 3;
#pragma unroll
    for (int r = 0; r < 8; ++r) {
      C[(     rr + r) * N + col] = acc0[r];
      C[(16 + rr + r) * N + col] = acc1[r];
      C[(32 + rr + r) * N + col] = acc2[r];
      C[(48 + rr + r) * N + col] = acc3[r];
    }
  }
}

// ---------------- embedding gather -> xs[S][B][512] bf16 ----------------
__global__ void embed_k(const int* __restrict__ text, const float* __restrict__ emb,
                        __bf16* __restrict__ xs) {
  size_t n = (size_t)NS * NB * 512;
  size_t st = (size_t)gridDim.x * blockDim.x;
  for (size_t i = (size_t)blockIdx.x * blockDim.x + threadIdx.x; i < n; i += st) {
    int d = (int)(i & 511), b = (int)((i >> 9) & 63), t = (int)(i >> 15);
    xs[i] = (__bf16)emb[(size_t)text[b * NS + t] * 512 + d];
  }
}

// ---------------- encoder LSTM scan: 1 workgroup per direction, h in LDS ----------------
// gates[t] = Xproj[t] (precomputed x@Wih^T + b) + h@Whh^T ; 4M x 1N blocked WMMA.
__global__ void __launch_bounds__(1024) lstm_scan(
    const float* __restrict__ Xf, const float* __restrict__ Xb,
    const __bf16* __restrict__ WhhF, const __bf16* __restrict__ WhhB,
    __bf16* __restrict__ xs_next, float* __restrict__ gg_all, float* __restrict__ c_all) {
  const int dir = blockIdx.x;
  const float* Xp = dir ? Xb : Xf;
  const __bf16* Whh = dir ? WhhB : WhhF;
  float* gg = gg_all + (size_t)dir * NB * 1024;
  float* cst = c_all + (size_t)dir * NB * 256;
  __shared__ __attribute__((aligned(32))) __bf16 hbuf[NB * 256];

  const int lane = threadIdx.x & 31, wv = threadIdx.x >> 5;
  const int acol = lane & 15;
  for (int i = threadIdx.x; i < NB * 256; i += 1024) { hbuf[i] = (__bf16)0.f; cst[i] = 0.f; }
  __syncthreads();

  for (int step = 0; step < NS; ++step) {
    const int t = dir ? (NS - 1 - step) : step;
    const float* Xt = Xp + (size_t)t * NB * 1024;
    // recurrent GEMM: gates[64x1024] = Xproj + h[64x256] @ Whh^T; 64 N-tiles over 32 waves
    for (int nt = wv; nt < 64; nt += 32) {
      const int n0 = nt << 4;
      const int col = n0 + acol;
      const int rr = (lane >> 4) << 3;
      v8f acc0, acc1, acc2, acc3;
#pragma unroll
      for (int r = 0; r < 8; ++r) {
        acc0[r] = Xt[(size_t)(     rr + r) * 1024 + col];
        acc1[r] = Xt[(size_t)(16 + rr + r) * 1024 + col];
        acc2[r] = Xt[(size_t)(32 + rr + r) * 1024 + col];
        acc3[r] = Xt[(size_t)(48 + rr + r) * 1024 + col];
      }
      for (int k0 = 0; k0 < 256; k0 += 32) {
        v16bf bf = load_frag_b(Whh, 256, n0 + acol, k0, lane);
        acc0 = wmma_bf16(load_frag_a(hbuf, 256,      acol, k0, lane), bf, acc0);
        acc1 = wmma_bf16(load_frag_a(hbuf, 256, 16 + acol, k0, lane), bf, acc1);
        acc2 = wmma_bf16(load_frag_a(hbuf, 256, 32 + acol, k0, lane), bf, acc2);
        acc3 = wmma_bf16(load_frag_a(hbuf, 256, 48 + acol, k0, lane), bf, acc3);
      }
#pragma unroll
      for (int r = 0; r < 8; ++r) {
        gg[(     rr + r) * 1024 + col] = acc0[r];
        gg[(16 + rr + r) * 1024 + col] = acc1[r];
        gg[(32 + rr + r) * 1024 + col] = acc2[r];
        gg[(48 + rr + r) * 1024 + col] = acc3[r];
      }
    }
    __syncthreads();
    // LSTM cell: gates order [i|f|g|o] chunks of 256
#pragma unroll
    for (int i = 0; i < 16; ++i) {
      int p = threadIdx.x + (i << 10);
      int b = p >> 8, j = p & 255;
      const float* g = gg + (b << 10);
      float cc = sigf(g[256 + j]) * cst[p] + sigf(g[j]) * tanh_f(g[512 + j]);
      cst[p] = cc;
      __bf16 hb = (__bf16)(sigf(g[768 + j]) * tanh_f(cc));
      hbuf[(b << 8) + j] = hb;
      xs_next[((size_t)t * NB + b) * 512 + dir * 256 + j] = hb;
    }
    __syncthreads();
  }
}

// ---------------- encoder finalize + decoder state init ----------------
__global__ void enc_finalize(const __bf16* __restrict__ xs3, float* __restrict__ encf,
                             __bf16* prev_in, float* ac, float* dc, float* dh,
                             __bf16* dh_bf, __bf16* dhin, unsigned* bar) {
  size_t st = (size_t)gridDim.x * blockDim.x;
  size_t t0 = (size_t)blockIdx.x * blockDim.x + threadIdx.x;
  for (size_t i = t0; i < (size_t)NB * NS * 512; i += st) {
    int d = (int)(i & 511), b = (int)((i >> 9) & 63), s = (int)(i >> 15);
    __bf16 v = xs3[i];
    encf[(((size_t)b * NS + s) << 9) + d] = (float)v;
    if (s == NS - 1) prev_in[b * AKP + 80 + d] = v;    // "last" columns
  }
  for (size_t i = t0; i < (size_t)NB * AKP; i += st) {
    int c = (int)(i % AKP);
    if (c < 80 || c >= 592) prev_in[i] = (__bf16)0.f;  // t=0 mel = 0, K padding = 0
  }
  for (size_t i = t0; i < (size_t)NB * DH; i += st) {
    ac[i] = 0.f; dc[i] = 0.f; dh[i] = 0.f; dh_bf[i] = (__bf16)0.f;
  }
  for (size_t i = t0; i < (size_t)NB * 1536; i += st) dhin[i] = (__bf16)0.f;
  if (t0 < 4) bar[t0] = 0u;
}

// ---------------- grid-wide barrier (sense-reversing, parity-split counters) ----------------
__device__ __forceinline__ void grid_sync(unsigned* bar) {
  __threadfence();
  __syncthreads();
  if (threadIdx.x == 0) {
    volatile unsigned* gen = bar + 2;
    unsigned g = *gen;
    unsigned idx = g & 1u;
    if (atomicAdd(bar + idx, 1u) == gridDim.x - 1) {
      bar[idx] = 0u;
      __threadfence();
      atomicAdd((unsigned*)(bar + 2), 1u);
    } else {
      while (*gen == g) __builtin_amdgcn_s_sleep(1);
    }
  }
  __syncthreads();
}

// ---------------- persistent decoder: 1000 steps, 8 grid barriers/step ----------------
__global__ void __launch_bounds__(256) decoder_kernel(
    const __bf16* __restrict__ aWih, const __bf16* __restrict__ aWhh, const float* __restrict__ a_b,
    const __bf16* __restrict__ wq, const float* __restrict__ we,
    const __bf16* __restrict__ dWih, const __bf16* __restrict__ dWhh, const float* __restrict__ d_b,
    const float* __restrict__ projW, const float* __restrict__ projb,
    const float* __restrict__ tmels, const float* __restrict__ enc, const __bf16* __restrict__ pk,
    float* __restrict__ out, __bf16* prev_in, __bf16* dhin, __bf16* dh_bf,
    float* ac, float* dc, float* dh, float* gates_a, float* gates_d,
    float* qbuf, float* scores, unsigned* bar) {
  const int lane = threadIdx.x & 31;
  const int wave = (blockIdx.x << 3) + (threadIdx.x >> 5);
  const int nwaves = gridDim.x << 3;
  const int tid = blockIdx.x * blockDim.x + threadIdx.x;
  const int nthr = gridDim.x * blockDim.x;

  for (int t = 0; t < NT; ++t) {
    // G1: attention-RNN gates = [mel|last] @ aWih^T + ah @ aWhh^T + b     (WMMA)
    mm64_tiles(prev_in, AKP, aWih, AKP, AKP, dhin, 1536, aWhh, 1024, 1024,
               a_b, gates_a, 4096, wave, nwaves, lane);
    grid_sync(bar);
    // G2: attention-RNN cell -> ah (bf16 into dhin[:, :1024]), ac
    for (int p = tid; p < NB * DH; p += nthr) {
      int b = p >> 10, j = p & 1023;
      const float* g = gates_a + (b << 12);
      float cc = sigf(g[1024 + j]) * ac[p] + sigf(g[j]) * tanh_f(g[2048 + j]);
      ac[p] = cc;
      dhin[b * 1536 + j] = (__bf16)(sigf(g[3072 + j]) * tanh_f(cc));
    }
    grid_sync(bar);
    // G3: q = ah @ Wq^T   (WMMA)
    mm64_tiles(dhin, 1536, wq, 1024, 1024, nullptr, 0, nullptr, 0, 0,
               nullptr, qbuf, 1024, wave, nwaves, lane);
    grid_sync(bar);
    // G4: scores[b][s] = sum_n tanh(q[b][n] + pk[s*64+b][n]) * We[n]  (trans-pipe)
    for (int p = wave; p < NB * NS; p += nwaves) {
      int b = p >> 9, s = p & 511;
      const __bf16* pkr = pk + ((size_t)(s * NB + b) << 10);
      const float* qr = qbuf + (b << 10);
      float sum = 0.f;
      for (int n = lane; n < 1024; n += 32)
        sum += tanh_f(qr[n] + (float)pkr[n]) * we[n];
      for (int o = 16; o; o >>= 1) sum += __shfl_down(sum, o, 32);
      if (lane == 0) scores[(b << 9) + s] = sum;
    }
    grid_sync(bar);
    // G5: softmax over s (one wave per batch row), in place
    if (wave < NB) {
      float* sr = scores + (wave << 9);
      float v[16], mx = -3.0e38f;
#pragma unroll
      for (int i = 0; i < 16; ++i) { v[i] = sr[lane + (i << 5)]; mx = fmaxf(mx, v[i]); }
      for (int o = 16; o; o >>= 1) mx = fmaxf(mx, __shfl_xor(mx, o, 32));
      float sm = 0.f;
#pragma unroll
      for (int i = 0; i < 16; ++i) { v[i] = __expf(v[i] - mx); sm += v[i]; }
      for (int o = 16; o; o >>= 1) sm += __shfl_xor(sm, o, 32);
      float inv = 1.f / sm;
#pragma unroll
      for (int i = 0; i < 16; ++i) sr[lane + (i << 5)] = v[i] * inv;
    }
    grid_sync(bar);
    // G5b: ctx[b][d] = sum_s w[b][s]*enc[b][s][d] -> dhin[:, 1024:1536] (bf16)
    for (int p = tid; p < NB * 512; p += nthr) {
      int b = p >> 9, d = p & 511;
      const float* wr = scores + (b << 9);
      const float* er = enc + ((size_t)b << 18) + d;
      float s = 0.f;
      for (int si = 0; si < 512; ++si) s += wr[si] * er[(size_t)si << 9];
      dhin[b * 1536 + 1024 + d] = (__bf16)s;
    }
    grid_sync(bar);
    // G6: decoder-RNN gates = [ah|ctx] @ dWih^T + dh @ dWhh^T + b   (WMMA)
    mm64_tiles(dhin, 1536, dWih, 1536, 1536, dh_bf, 1024, dWhh, 1024, 1024,
               d_b, gates_d, 4096, wave, nwaves, lane);
    grid_sync(bar);
    // G7: decoder-RNN cell + stage next step's prev-mel (teacher forcing)
    for (int p = tid; p < NB * DH; p += nthr) {
      int b = p >> 10, j = p & 1023;
      const float* g = gates_d + (b << 12);
      float cc = sigf(g[1024 + j]) * dc[p] + sigf(g[j]) * tanh_f(g[2048 + j]);
      dc[p] = cc;
      float hh = sigf(g[3072 + j]) * tanh_f(cc);
      dh[p] = hh;
      dh_bf[p] = (__bf16)hh;
    }
    for (int p = tid; p < NB * NMEL; p += nthr) {
      int b = p / NMEL, m = p - b * NMEL;
      prev_in[b * AKP + m] = (__bf16)tmels[((size_t)b * NT + t) * NMEL + m];
    }
    grid_sync(bar);
    // G8: mel projection -> out (no trailing barrier: next G1 touches nothing G8 reads/writes)
    for (int p = tid; p < NB * NMEL; p += nthr) {
      int b = p / NMEL, m = p - b * NMEL;
      const float* wr = projW + (m << 10);
      const float* hr = dh + (b << 10);
      float s = projb[m];
      for (int j = 0; j < 1024; ++j) s += wr[j] * hr[j];
      out[((size_t)b * NT + t) * NMEL + m] = s;
    }
  }
}

// ---------------- postnet convs (f32 VALU; <10% of total FLOPs) ----------------
__global__ void conv1_k(const float* __restrict__ mels, const float* __restrict__ w,
                        const float* __restrict__ bias, const float* __restrict__ g,
                        const float* __restrict__ bb, const float* __restrict__ mean,
                        const float* __restrict__ var, float* __restrict__ out) {
  size_t n = (size_t)NB * 512 * NT;
  size_t st = (size_t)gridDim.x * blockDim.x;
  for (size_t i = (size_t)blockIdx.x * blockDim.x + threadIdx.x; i < n; i += st) {
    int t = (int)(i % NT); int oc = (int)((i / NT) & 511); int b = (int)(i / ((size_t)NT * 512));
    float s = bias[oc];
#pragma unroll
    for (int k = 0; k < 5; ++k) {
      int tt = t + k - 2;
      if ((unsigned)tt < (unsigned)NT) {
        const float* mr = mels + ((size_t)b * NT + tt) * NMEL;
        const float* wr = w + (size_t)oc * NMEL * 5 + k;
        for (int ic = 0; ic < NMEL; ++ic) s += wr[ic * 5] * mr[ic];
      }
    }
    float y = (s - mean[oc]) * rsqrtf(var[oc] + 1e-5f) * g[oc] + bb[oc];
    out[i] = tanh_f(y);
  }
}
__global__ void conv2_k(const float* __restrict__ in, const float* __restrict__ w,
                        const float* __restrict__ bias, const float* __restrict__ g,
                        const float* __restrict__ bb, const float* __restrict__ mean,
                        const float* __restrict__ var, float* __restrict__ out) {
  size_t n = (size_t)NB * 512 * NT;
  size_t st = (size_t)gridDim.x * blockDim.x;
  for (size_t i = (size_t)blockIdx.x * blockDim.x + threadIdx.x; i < n; i += st) {
    int t = (int)(i % NT); int oc = (int)((i / NT) & 511); int b = (int)(i / ((size_t)NT * 512));
    const float* inb = in + (size_t)b * 512 * NT;
    float s = bias[oc];
    for (int ic = 0; ic < 512; ++ic) {
      const float* row = inb + (size_t)ic * NT;
      const float* wr = w + ((size_t)oc * 512 + ic) * 5;
#pragma unroll
      for (int k = 0; k < 5; ++k) {
        int tt = t + k - 2;
        if ((unsigned)tt < (unsigned)NT) s += wr[k] * row[tt];
      }
    }
    float y = (s - mean[oc]) * rsqrtf(var[oc] + 1e-5f) * g[oc] + bb[oc];
    out[i] = tanh_f(y);
  }
}
__global__ void conv3_k(const float* __restrict__ in, const float* __restrict__ w,
                        const float* __restrict__ bias, float* __restrict__ out) {
  size_t n = (size_t)NB * NMEL * NT;
  size_t st = (size_t)gridDim.x * blockDim.x;
  for (size_t i = (size_t)blockIdx.x * blockDim.x + threadIdx.x; i < n; i += st) {
    int t = (int)(i % NT); int m = (int)((i / NT) % NMEL); int b = (int)(i / ((size_t)NT * NMEL));
    const float* inb = in + (size_t)b * 512 * NT;
    float s = bias[m];
    for (int ic = 0; ic < 512; ++ic) {
      const float* row = inb + (size_t)ic * NT;
      const float* wr = w + ((size_t)m * 512 + ic) * 5;
#pragma unroll
      for (int k = 0; k < 5; ++k) {
        int tt = t + k - 2;
        if ((unsigned)tt < (unsigned)NT) s += wr[k] * row[tt];
      }
    }
    out[((size_t)b * NT + t) * NMEL + m] += s;  // residual: mels already in out
  }
}

// ---------------- host ----------------
extern "C" void kernel_launch(void* const* d_in, const int* in_sizes, int n_in,
                              void* d_out, int out_size, void* d_ws, size_t ws_size,
                              hipStream_t stream) {
  (void)in_sizes; (void)n_in; (void)out_size; (void)ws_size;
  const int*   text   = (const int*)d_in[0];
  const float* tmels  = (const float*)d_in[1];
  const float* emb    = (const float*)d_in[2];
  const float* eWih   = (const float*)d_in[3];
  const float* eWhh   = (const float*)d_in[4];
  const float* e_b    = (const float*)d_in[5];
  const float* aWih   = (const float*)d_in[6];
  const float* aWhh   = (const float*)d_in[7];
  const float* a_b    = (const float*)d_in[8];
  const float* Wq     = (const float*)d_in[9];
  const float* Wk     = (const float*)d_in[10];
  const float* We     = (const float*)d_in[11];
  const float* dWih   = (const float*)d_in[12];
  const float* dWhh   = (const float*)d_in[13];
  const float* d_b    = (const float*)d_in[14];
  const float* projW  = (const float*)d_in[15];
  const float* projb  = (const float*)d_in[16];
  const float* pW1 = (const float*)d_in[17]; const float* pb1 = (const float*)d_in[18];
  const float* g1 = (const float*)d_in[19]; const float* b1 = (const float*)d_in[20];
  const float* m1 = (const float*)d_in[21]; const float* v1 = (const float*)d_in[22];
  const float* pW2 = (const float*)d_in[23]; const float* pb2 = (const float*)d_in[24];
  const float* g2 = (const float*)d_in[25]; const float* b2 = (const float*)d_in[26];
  const float* m2 = (const float*)d_in[27]; const float* v2 = (const float*)d_in[28];
  const float* pW3 = (const float*)d_in[29]; const float* pb3 = (const float*)d_in[30];
  float* out = (float*)d_out;

  // workspace layout (~520 MB; Xf/Xb reused as postnet temps)
  size_t off = 0;
  char* ws = (char*)d_ws;
  auto take = [&](size_t bytes) -> void* {
    void* p = ws + off; off += (bytes + 255) & ~(size_t)255; return p;
  };
  __bf16* eWih_bf = (__bf16*)take((size_t)3 * 2 * 1024 * 512 * 2);
  __bf16* eWhh_bf = (__bf16*)take((size_t)3 * 2 * 1024 * 256 * 2);
  __bf16* aWih_bf = (__bf16*)take((size_t)4096 * AKP * 2);
  __bf16* aWhh_bf = (__bf16*)take((size_t)4096 * 1024 * 2);
  __bf16* wq_bf   = (__bf16*)take((size_t)1024 * 1024 * 2);
  __bf16* wk_bf   = (__bf16*)take((size_t)1024 * 512 * 2);
  __bf16* dWih_bf = (__bf16*)take((size_t)4096 * 1536 * 2);
  __bf16* dWhh_bf = (__bf16*)take((size_t)4096 * 1024 * 2);
  __bf16* xsA  = (__bf16*)take((size_t)NS * NB * 512 * 2);
  __bf16* xsB  = (__bf16*)take((size_t)NS * NB * 512 * 2);
  float*  Xf   = (float*)take((size_t)NS * NB * 1024 * 4);   // also postnet p1
  float*  Xb   = (float*)take((size_t)NS * NB * 1024 * 4);   // also postnet p2
  float*  ggbuf = (float*)take((size_t)2 * NB * 1024 * 4);
  float*  cstate = (float*)take((size_t)2 * NB * 256 * 4);
  float*  encf = (float*)take((size_t)NB * NS * 512 * 4);
  __bf16* pk   = (__bf16*)take((size_t)NS * NB * 1024 * 2);
  float*  ac = (float*)take((size_t)NB * DH * 4);
  float*  dc = (float*)take((size_t)NB * DH * 4);
  float*  dh = (float*)take((size_t)NB * DH * 4);
  __bf16* dh_bf = (__bf16*)take((size_t)NB * DH * 2);
  __bf16* dhin  = (__bf16*)take((size_t)NB * 1536 * 2);
  __bf16* prev_in = (__bf16*)take((size_t)NB * AKP * 2);
  float*  gates_a = (float*)take((size_t)NB * 4096 * 4);
  float*  gates_d = (float*)take((size_t)NB * 4096 * 4);
  float*  qbuf   = (float*)take((size_t)NB * 1024 * 4);
  float*  scores = (float*)take((size_t)NB * 512 * 4);
  unsigned* bar  = (unsigned*)take(256);

  // 1) weights fp32 -> bf16 (L2-resident working set, ~47 MB)
  f2bf<<<1024, 256, 0, stream>>>(eWih, eWih_bf, (size_t)3 * 2 * 1024 * 512);
  f2bf<<<1024, 256, 0, stream>>>(eWhh, eWhh_bf, (size_t)3 * 2 * 1024 * 256);
  f2bf_pad<<<1024, 256, 0, stream>>>(aWih, aWih_bf, 4096, 592, AKP);
  f2bf<<<1024, 256, 0, stream>>>(aWhh, aWhh_bf, (size_t)4096 * 1024);
  f2bf<<<1024, 256, 0, stream>>>(Wq, wq_bf, (size_t)1024 * 1024);
  f2bf<<<1024, 256, 0, stream>>>(Wk, wk_bf, (size_t)1024 * 512);
  f2bf<<<1024, 256, 0, stream>>>(dWih, dWih_bf, (size_t)4096 * 1536);
  f2bf<<<1024, 256, 0, stream>>>(dWhh, dWhh_bf, (size_t)4096 * 1024);

  // 2) embedding -> xs0 [S][B][512] bf16
  embed_k<<<4096, 256, 0, stream>>>(text, emb, xsA);

  // 3) encoder: 3 bi-LSTM layers; input projections hoisted to big WMMA GEMMs
  __bf16* cur = xsA; __bf16* nxt = xsB;
  for (int l = 0; l < 3; ++l) {
    gemm_bf16<<<2048, 256, 0, stream>>>(cur, 512, eWih_bf + (size_t)(l * 2 + 0) * 1024 * 512, 512,
                                        Xf, nullptr, 1024, e_b + (l * 2 + 0) * 1024,
                                        NS * NB, 1024, 512);
    gemm_bf16<<<2048, 256, 0, stream>>>(cur, 512, eWih_bf + (size_t)(l * 2 + 1) * 1024 * 512, 512,
                                        Xb, nullptr, 1024, e_b + (l * 2 + 1) * 1024,
                                        NS * NB, 1024, 512);
    lstm_scan<<<2, 1024, 0, stream>>>(Xf, Xb,
                                      eWhh_bf + (size_t)(l * 2 + 0) * 1024 * 256,
                                      eWhh_bf + (size_t)(l * 2 + 1) * 1024 * 256,
                                      nxt, ggbuf, cstate);
    __bf16* tmp = cur; cur = nxt; nxt = tmp;
  }

  // 4) finalize encoder (enc f32, "last" into prev_in) + zero decoder state + barrier
  enc_finalize<<<4096, 256, 0, stream>>>(cur, encf, prev_in, ac, dc, dh, dh_bf, dhin, bar);

  // 5) pk = enc @ Wk^T, stored bf16 (64 MB, read 1000x mostly from L2)
  gemm_bf16<<<2048, 256, 0, stream>>>(cur, 512, wk_bf, 512, nullptr, pk, 1024,
                                      nullptr, NS * NB, 1024, 512);

  // 6) persistent decoder: 1000 steps, WMMA gates + attention, mels -> d_out
  decoder_kernel<<<128, 256, 0, stream>>>(aWih_bf, aWhh_bf, a_b, wq_bf, We,
                                          dWih_bf, dWhh_bf, d_b, projW, projb,
                                          tmels, encf, pk, out,
                                          prev_in, dhin, dh_bf, ac, dc, dh,
                                          gates_a, gates_d, qbuf, scores, bar);

  // 7) postnet: conv1/conv2/conv3, residual added into d_out
  conv1_k<<<8192, 256, 0, stream>>>(out, pW1, pb1, g1, b1, m1, v1, Xf);
  conv2_k<<<8192, 256, 0, stream>>>(Xf, pW2, pb2, g2, b2, m2, v2, Xb);
  conv3_k<<<2048, 256, 0, stream>>>(Xb, pW3, pb3, out);
}
